// GraphormerMultiHeadAttention_45844480918017
// MI455X (gfx1250) — compile-verified, matrix operand order checked by reading, not run on hardware
//
#include <hip/hip_runtime.h>
#include <hip/hip_bf16.h>

#define NUM_HEADS 12
#define HIDDEN    768
#define HEAD      64
#define BATCH     16
#define SEQ       512
#define TOK       (BATCH * SEQ)   // 8192
#define AST       40              // padded LDS row stride (halves) for 32-half rows

typedef __attribute__((ext_vector_type(16))) _Float16 v16h;
typedef __attribute__((ext_vector_type(8)))  float    v8f;

union F16Frag {
    v16h     v;
    _Float16 h[16];
    uint4    q[2];
};

__device__ __forceinline__ v8f wmma_f16f32(v16h a, v16h b, v8f c) {
    // D = A(16x32 f16) x B(32x16 f16) + C(16x16 f32)
    return __builtin_amdgcn_wmma_f32_16x16x32_f16(
        /*neg_a=*/false, a, /*neg_b=*/false, b,
        /*c_mod=*/(short)0, c, /*reuse_a=*/false, /*reuse_b=*/false);
}

// --- CDNA5 async copy: global -> LDS, 16B per lane, ASYNCcnt-tracked --------
__device__ __forceinline__ void async_b128(const void* gsrc, void* ldst) {
    // generic shared pointer: low 32 bits are the LDS byte address (ISA 10.2)
    unsigned loff = (unsigned)(unsigned long long)ldst;
    asm volatile("global_load_async_to_lds_b128 %0, %1, off"
                 :: "v"(loff), "v"(gsrc) : "memory");
}
__device__ __forceinline__ void wait_async0() {
    asm volatile("s_wait_asynccnt 0x0" ::: "memory");
}
__device__ __forceinline__ void wait_async4() {
    asm volatile("s_wait_asynccnt 0x4" ::: "memory");
}

// ---------------------------------------------------------------------------
// f32 -> f16 conversion (vectorized x4)
// ---------------------------------------------------------------------------
__global__ void cvt_f32_to_f16(const float* __restrict__ in,
                               _Float16* __restrict__ out, int n4) {
    int i = blockIdx.x * blockDim.x + threadIdx.x;
    if (i >= n4) return;
    float4 f = ((const float4*)in)[i];
    union { _Float16 h[4]; uint2 u; } o;
    o.h[0] = (_Float16)f.x; o.h[1] = (_Float16)f.y;
    o.h[2] = (_Float16)f.z; o.h[3] = (_Float16)f.w;
    ((uint2*)out)[i] = o.u;
}

// ---------------------------------------------------------------------------
// C(MxN) = A(MxK,f16) @ W(NxK,f16)^T [+ bias]   (torch Linear convention)
// 256 threads = 8 waves in 4(M) x 2(N); block tile 128x128, K-step 32.
// Double-buffered LDS staged with async global->LDS copies; each wave does a
// 32x64 sub-tile = 8 v_wmma per K-step with B-fragment reuse.
// ---------------------------------------------------------------------------
template <bool OUT_F32>
__global__ __launch_bounds__(256) void gemm_bt(
    const _Float16* __restrict__ A, const _Float16* __restrict__ W,
    const float* __restrict__ bias, void* __restrict__ Cout,
    int M, int N, int K) {
    __shared__ __align__(16) _Float16 lA[2][128 * AST];
    __shared__ __align__(16) _Float16 lB[2][128 * AST];

    const int tid  = threadIdx.x;
    const int wave = tid >> 5;
    const int lane = tid & 31;
    const int lrow = lane & 15;
    const int lh   = lane >> 4;
    const int wm   = wave >> 1;   // 0..3 : 32-row strip
    const int wn   = wave & 1;    // 0..1 : 64-col strip
    const int m0   = blockIdx.x * 128;
    const int n0   = blockIdx.y * 128;

    v8f acc[2][4];
#pragma unroll
    for (int i = 0; i < 2; ++i)
#pragma unroll
        for (int t = 0; t < 4; ++t)
#pragma unroll
            for (int v = 0; v < 8; ++v) acc[i][t][v] = 0.f;

    // staging assignment: thread covers 32B of A and 32B of W per tile
    const int r  = tid >> 1;          // 0..127
    const int c0 = (tid & 1) << 4;    // 0 or 16 (halves)

    auto stage = [&](int buf, int k0) {
        const _Float16* ga = A + (size_t)(m0 + r) * K + k0;
        const _Float16* gw = W + (size_t)(n0 + r) * K + k0;
        async_b128(ga + c0,     &lA[buf][r * AST + c0]);
        async_b128(ga + c0 + 8, &lA[buf][r * AST + c0 + 8]);
        async_b128(gw + c0,     &lB[buf][r * AST + c0]);
        async_b128(gw + c0 + 8, &lB[buf][r * AST + c0 + 8]);
    };

    stage(0, 0);
    int cur = 0;
    for (int k0 = 0; k0 < K; k0 += 32) {
        const bool more = (k0 + 32) < K;
        if (more) stage(cur ^ 1, k0 + 32);   // prefetch next tile
        if (more) wait_async4(); else wait_async0();  // current tile landed
        __syncthreads();

        F16Frag af[2];
#pragma unroll
        for (int i = 0; i < 2; ++i) {
            const _Float16* ap = &lA[cur][(wm * 32 + i * 16 + lrow) * AST + lh * 8];
            af[i].q[0] = *(const uint4*)(ap);
            af[i].q[1] = *(const uint4*)(ap + 16);
        }
#pragma unroll
        for (int t = 0; t < 4; ++t) {
            F16Frag bf;
            const _Float16* bp = &lB[cur][(wn * 64 + t * 16 + lrow) * AST + lh * 8];
            bf.q[0] = *(const uint4*)(bp);
            bf.q[1] = *(const uint4*)(bp + 16);
            acc[0][t] = wmma_f16f32(af[0].v, bf.v, acc[0][t]);
            acc[1][t] = wmma_f16f32(af[1].v, bf.v, acc[1][t]);
        }
        __syncthreads();   // all waves done reading before buffer reuse
        cur ^= 1;
    }

#pragma unroll
    for (int i = 0; i < 2; ++i)
#pragma unroll
        for (int t = 0; t < 4; ++t) {
            const int   n    = n0 + wn * 64 + t * 16 + lrow;
            const float bval = bias ? bias[n] : 0.f;
#pragma unroll
            for (int v = 0; v < 8; ++v) {
                const int   m   = m0 + wm * 32 + i * 16 + v + lh * 8;
                const float val = acc[i][t][v] + bval;
                if (OUT_F32)
                    ((float*)Cout)[(size_t)m * N + n] = val;
                else
                    ((_Float16*)Cout)[(size_t)m * N + n] = (_Float16)val;
            }
        }
}

// ---------------------------------------------------------------------------
// Fused attention: 4 waves per block, each owns a 16-query tile of one
// (batch, head); the 32-key V chunk is async-staged into LDS once per block
// and shared. Flash-style online softmax; scores never touch global memory.
// ---------------------------------------------------------------------------
__global__ __launch_bounds__(128) void attn_fa(
    const _Float16* __restrict__ Qh, const _Float16* __restrict__ Kh,
    const _Float16* __restrict__ Vh, const float* __restrict__ spatial,
    const float* __restrict__ edge, _Float16* __restrict__ Oh) {
    __shared__ __align__(16) _Float16 lP[4][16 * AST];
    __shared__ __align__(16) _Float16 lV[32 * 64];

    const int tid  = threadIdx.x;
    const int wave = tid >> 5;
    const int lane = tid & 31;
    const int lrow = lane & 15;
    const int lh   = lane >> 4;
    const int q0   = blockIdx.x * 64 + wave * 16;
    const int h    = blockIdx.y;
    const int b    = blockIdx.z;
    const float scale = 0.03608439182435161f;  // HIDDEN^-0.5 (not head^-0.5)

    const size_t    tokbase = (size_t)b * SEQ;
    const _Float16* Qbase   = Qh + (tokbase + q0) * HIDDEN + h * HEAD;

    F16Frag qf[2];
#pragma unroll
    for (int c = 0; c < 2; ++c) {
        const _Float16* p = Qbase + (size_t)lrow * HIDDEN + c * 32 + lh * 8;
        qf[c].q[0] = *(const uint4*)(p);
        qf[c].q[1] = *(const uint4*)(p + 16);
    }

    v8f   oacc[4];
    float rmax[8], rsum[8], rnz[8];
#pragma unroll
    for (int v = 0; v < 8; ++v) {
        rmax[v] = -__builtin_inff(); rsum[v] = 0.f; rnz[v] = 0.f;
    }
#pragma unroll
    for (int t = 0; t < 4; ++t)
#pragma unroll
        for (int v = 0; v < 8; ++v) oacc[t][v] = 0.f;

    const float* spq = spatial + ((size_t)b * SEQ + q0) * SEQ;
    const float* edq = edge    + ((size_t)b * SEQ + q0) * SEQ;

    const int vr = tid >> 2;          // 0..31 : V row (key)
    const int vc = (tid & 3) << 4;    // 0/16/32/48 (halves)

    for (int k0 = 0; k0 < SEQ; k0 += 32) {
        // (a) kick off async V staging for this chunk (overlaps QK^T)
        {
            const _Float16* vsrc =
                Vh + (tokbase + k0 + vr) * HIDDEN + h * HEAD + vc;
            async_b128(vsrc,     lV + vr * 64 + vc);
            async_b128(vsrc + 8, lV + vr * 64 + vc + 8);
        }

        // (b) S = Q @ K^T  (16 x 32 chunk, fp32 acc)
        const _Float16* Kbase = Kh + (tokbase + k0) * HIDDEN + h * HEAD;
        v8f s[2];
#pragma unroll
        for (int t = 0; t < 2; ++t)
#pragma unroll
            for (int v = 0; v < 8; ++v) s[t][v] = 0.f;
#pragma unroll
        for (int t = 0; t < 2; ++t)
#pragma unroll
            for (int c = 0; c < 2; ++c) {
                F16Frag kf;
                const _Float16* p =
                    Kbase + (size_t)(t * 16 + lrow) * HIDDEN + c * 32 + lh * 8;
                kf.q[0] = *(const uint4*)(p);
                kf.q[1] = *(const uint4*)(p + 16);
                s[t]    = wmma_f16f32(qf[c].v, kf.v, s[t]);
            }

        // (c) scale + graph bias + zero-row tracking
#pragma unroll
        for (int t = 0; t < 2; ++t)
#pragma unroll
            for (int v = 0; v < 8; ++v) {
                const int   m  = v + lh * 8;
                const int   n  = k0 + t * 16 + lrow;
                const float bb = spq[(size_t)m * SEQ + n] + edq[(size_t)m * SEQ + n];
                const float sv = s[t][v] * scale + bb;
                s[t][v] = sv;
                rnz[v]  = fmaxf(rnz[v], (sv != 0.f) ? 1.f : 0.f);
            }

        // (d) online softmax (row lives across 16 lanes of a half-wave)
        float corr[8];
#pragma unroll
        for (int v = 0; v < 8; ++v) {
            float mx = fmaxf(s[0][v], s[1][v]);
#pragma unroll
            for (int d = 8; d >= 1; d >>= 1)
                mx = fmaxf(mx, __shfl_xor(mx, d, 32));
            const float mnew = fmaxf(rmax[v], mx);
            const float cf   = __expf(rmax[v] - mnew);
            rmax[v] = mnew;
            const float p0 = __expf(s[0][v] - mnew);
            const float p1 = __expf(s[1][v] - mnew);
            s[0][v] = p0; s[1][v] = p1;
            float rs = p0 + p1;
#pragma unroll
            for (int d = 8; d >= 1; d >>= 1) rs += __shfl_xor(rs, d, 32);
            rsum[v] = rsum[v] * cf + rs;
            corr[v] = cf;
        }
#pragma unroll
        for (int t = 0; t < 4; ++t)
#pragma unroll
            for (int v = 0; v < 8; ++v) oacc[t][v] *= corr[v];

        // (e) P (C-layout) -> LDS, re-read as A-fragment
        _Float16* lPw = lP[wave];
#pragma unroll
        for (int t = 0; t < 2; ++t)
#pragma unroll
            for (int v = 0; v < 8; ++v)
                lPw[(v + lh * 8) * AST + t * 16 + lrow] = (_Float16)s[t][v];

        wait_async0();      // own async V copies done
        __syncthreads();    // everyone's copies + lP visible

        F16Frag pf;
        pf.q[0] = *(const uint4*)(lPw + lrow * AST + lh * 8);
        pf.q[1] = *(const uint4*)(lPw + lrow * AST + lh * 8 + 16);

#pragma unroll
        for (int t = 0; t < 4; ++t) {
            F16Frag vf;
#pragma unroll
            for (int i = 0; i < 8; ++i) {
                vf.h[i]     = lV[(lh * 8 + i) * 64      + t * 16 + lrow];
                vf.h[8 + i] = lV[(16 + lh * 8 + i) * 64 + t * 16 + lrow];
            }
            oacc[t] = wmma_f16f32(pf.v, vf.v, oacc[t]);
        }
        __syncthreads();    // before lV/lP reuse next iteration
    }

    // finalize: 1/rowsum, zero rows whose scores were all exactly zero
    float fin[8];
#pragma unroll
    for (int v = 0; v < 8; ++v) {
        float f = rnz[v];
#pragma unroll
        for (int d = 8; d >= 1; d >>= 1) f = fmaxf(f, __shfl_xor(f, d, 32));
        fin[v] = (f != 0.f) ? (1.f / rsum[v]) : 0.f;
    }
    _Float16* Obase = Oh + (tokbase + q0) * HIDDEN + h * HEAD;
#pragma unroll
    for (int t = 0; t < 4; ++t)
#pragma unroll
        for (int v = 0; v < 8; ++v)
            Obase[(size_t)(v + lh * 8) * HIDDEN + t * 16 + lrow] =
                (_Float16)(oacc[t][v] * fin[v]);
}

// ---------------------------------------------------------------------------
extern "C" void kernel_launch(void* const* d_in, const int* in_sizes, int n_in,
                              void* d_out, int out_size, void* d_ws,
                              size_t ws_size, hipStream_t stream) {
    (void)in_sizes; (void)n_in; (void)out_size; (void)ws_size;
    const float* x  = (const float*)d_in[0];
    const float* sp = (const float*)d_in[1];
    const float* ed = (const float*)d_in[2];
    const float* Wq = (const float*)d_in[3];
    const float* Wk = (const float*)d_in[4];
    const float* Wv = (const float*)d_in[5];
    const float* bv = (const float*)d_in[6];
    const float* Wo = (const float*)d_in[7];
    const float* bo = (const float*)d_in[8];
    float*       out = (float*)d_out;

    _Float16* xh  = (_Float16*)d_ws;
    _Float16* wqh = xh  + (size_t)TOK * HIDDEN;
    _Float16* wkh = wqh + (size_t)HIDDEN * HIDDEN;
    _Float16* wvh = wkh + (size_t)HIDDEN * HIDDEN;
    _Float16* woh = wvh + (size_t)HIDDEN * HIDDEN;
    _Float16* Qh  = woh + (size_t)HIDDEN * HIDDEN;
    _Float16* Kh  = Qh  + (size_t)TOK * HIDDEN;
    _Float16* Vh  = Kh  + (size_t)TOK * HIDDEN;
    _Float16* Oh  = Vh  + (size_t)TOK * HIDDEN;

    const int n4 = TOK * HIDDEN / 4;
    const int w4 = HIDDEN * HIDDEN / 4;
    cvt_f32_to_f16<<<(n4 + 255) / 256, 256, 0, stream>>>(x, xh, n4);
    cvt_f32_to_f16<<<(w4 + 255) / 256, 256, 0, stream>>>(Wq, wqh, w4);
    cvt_f32_to_f16<<<(w4 + 255) / 256, 256, 0, stream>>>(Wk, wkh, w4);
    cvt_f32_to_f16<<<(w4 + 255) / 256, 256, 0, stream>>>(Wv, wvh, w4);
    cvt_f32_to_f16<<<(w4 + 255) / 256, 256, 0, stream>>>(Wo, woh, w4);

    dim3 gg(TOK / 128, HIDDEN / 128);
    gemm_bt<false><<<gg, 256, 0, stream>>>(xh, wqh, nullptr, Qh, TOK, HIDDEN, HIDDEN);
    gemm_bt<false><<<gg, 256, 0, stream>>>(xh, wkh, nullptr, Kh, TOK, HIDDEN, HIDDEN);
    gemm_bt<false><<<gg, 256, 0, stream>>>(xh, wvh, bv,      Vh, TOK, HIDDEN, HIDDEN);

    dim3 ga(SEQ / 64, NUM_HEADS, BATCH);
    attn_fa<<<ga, 128, 0, stream>>>(Qh, Kh, Vh, sp, ed, Oh);

    gemm_bt<true><<<gg, 256, 0, stream>>>(Oh, woh, bo, out, TOK, HIDDEN, HIDDEN);
}
